// LayoutGraphModel_30124900614423
// MI455X (gfx1250) — matrix operand order repaired and checked by reading, not compiled, not run on hardware
//
#include <hip/hip_runtime.h>
#include <hip/hip_bf16.h>

typedef __attribute__((ext_vector_type(16))) _Float16 v16h;
typedef __attribute__((ext_vector_type(8)))  _Float16 v8h;
typedef __attribute__((ext_vector_type(8)))  float    v8f;

// ---------------------------------------------------------------------------
// seg_id[i] = searchsorted(sep, i, 'right')
__global__ void k_segid(const int* __restrict__ sep, int S, int N, int* __restrict__ segid) {
    int i = blockIdx.x * blockDim.x + threadIdx.x;
    if (i >= N) return;
    int lo = 0, hi = S;
    while (lo < hi) { int mid = (lo + hi) >> 1; if (sep[mid] <= i) lo = mid + 1; else hi = mid; }
    segid[i] = lo;
}

// W [Din,Dout] f32  ->  WT [DoutP][DinP] f16 (transposed, zero padded)
__global__ void k_convWT(const float* __restrict__ W, _Float16* __restrict__ WT,
                         int Din, int Dout, int DinP, int DoutP) {
    int t = blockIdx.x * blockDim.x + threadIdx.x;
    if (t >= DoutP * DinP) return;
    int n = t / DinP, k = t % DinP;
    float v = (n < Dout && k < Din) ? W[(size_t)k * Dout + n] : 0.0f;
    WT[t] = (_Float16)v;
}

// Assemble x0 [N,192] f16: [cat_nf(16) | cat_cf(8) | num(101) | op_emb(64) | pad(3)]
__global__ void k_build_x0(const float* __restrict__ node_features,      // [N,105]
                           const int*   __restrict__ node_config_feat,   // [N,2]
                           const int*   __restrict__ node_ops,           // [N]
                           const float* __restrict__ opcode_emb,         // [120,64]
                           const float* __restrict__ cat_emb,            // [7,4]
                           _Float16* __restrict__ x0, int N) {
    int t = blockIdx.x * blockDim.x + threadIdx.x;
    if (t >= N * 192) return;
    int i = t / 192, c = t % 192;
    float v = 0.0f;
    if (c < 16) {
        int j = c >> 2, d = c & 3;
        int idx = (int)node_features[(size_t)i * 105 + 101 + j];
        v = cat_emb[idx * 4 + d];
    } else if (c < 24) {
        int cc = c - 16; int j = cc >> 2, d = cc & 3;
        int idx = node_config_feat[(size_t)i * 2 + j];
        v = cat_emb[idx * 4 + d];
    } else if (c < 125) {
        v = node_features[(size_t)i * 105 + (c - 24)];
    } else if (c < 189) {
        v = opcode_emb[(size_t)node_ops[i] * 64 + (c - 125)];
    }
    x0[(size_t)i * 192 + c] = (_Float16)v;
}

// ---------------------------------------------------------------------------
// Register-blocked wave-tile WMMA GEMM: each wave owns a 16x64 output tile
// (4 accumulators), reusing one A-fragment across 4 B column tiles per K-step.
// C[M,Nout] = A1(f16)@WT1^T (+ (A2(f32)*rowScale)@WT2^T) + bias
// A row-major stride Kp; WT row-major [Nout][Kp].
// EPI 0: v = leaky(v + bias), zero pad cols >= realN.
// EPI 1: v = v + bias, store f16, atomicAdd rownorm[row] += v*v.
template<int EPI>
__global__ void gemm_wmma(const _Float16* __restrict__ A1,
                          const _Float16* __restrict__ WT1,
                          const float*    __restrict__ A2,
                          const _Float16* __restrict__ WT2,
                          const float*    __restrict__ rowScale,
                          const float*    __restrict__ bias,
                          _Float16* __restrict__ Cc,
                          float* __restrict__ rownorm,
                          int M, int Kp, int Nout, int realN) {
    int tid  = threadIdx.x;
    int wave = tid >> 5;
    int lane = tid & 31;
    int hl   = lane & 15;
    int hi   = lane >> 4;
    int m0 = blockIdx.x * 128 + wave * 16;
    int n0 = blockIdx.y * 64;
    if (m0 >= M) return;

    v8f acc0 = {}, acc1 = {}, acc2 = {}, acc3 = {};
    int rowA = m0 + hl;

    {   // pass 1: f16 A
        const _Float16* aBase = A1 + (size_t)rowA * Kp;
        const _Float16* bB0 = WT1 + (size_t)(n0 + 0 * 16 + hl) * Kp;
        const _Float16* bB1 = WT1 + (size_t)(n0 + 1 * 16 + hl) * Kp;
        const _Float16* bB2 = WT1 + (size_t)(n0 + 2 * 16 + hl) * Kp;
        const _Float16* bB3 = WT1 + (size_t)(n0 + 3 * 16 + hl) * Kp;
        for (int k0 = 0; k0 < Kp; k0 += 32) {
            v8h a_lo = *(const v8h*)(aBase + k0 + hi * 8);
            v8h a_hi = *(const v8h*)(aBase + k0 + 16 + hi * 8);
            v16h av;
#pragma unroll
            for (int i = 0; i < 8; ++i) { av[i] = a_lo[i]; av[8 + i] = a_hi[i]; }
#pragma unroll
            for (int j = 0; j < 4; ++j) {
                const _Float16* bB = (j == 0) ? bB0 : (j == 1) ? bB1 : (j == 2) ? bB2 : bB3;
                v8h b_lo = *(const v8h*)(bB + k0 + hi * 8);
                v8h b_hi = *(const v8h*)(bB + k0 + 16 + hi * 8);
                v16h bv;
#pragma unroll
                for (int i = 0; i < 8; ++i) { bv[i] = b_lo[i]; bv[8 + i] = b_hi[i]; }
                v8f& acc = (j == 0) ? acc0 : (j == 1) ? acc1 : (j == 2) ? acc2 : acc3;
                acc = __builtin_amdgcn_wmma_f32_16x16x32_f16(false, av, false, bv,
                                                             (short)0, acc, false, false);
            }
        }
    }
    if (A2) {  // pass 2: f32 aggregate scaled by invdeg, converted in-register
        float sc = rowScale ? rowScale[rowA] : 1.0f;
        const float* aBase = A2 + (size_t)rowA * Kp;
        const _Float16* bB0 = WT2 + (size_t)(n0 + 0 * 16 + hl) * Kp;
        const _Float16* bB1 = WT2 + (size_t)(n0 + 1 * 16 + hl) * Kp;
        const _Float16* bB2 = WT2 + (size_t)(n0 + 2 * 16 + hl) * Kp;
        const _Float16* bB3 = WT2 + (size_t)(n0 + 3 * 16 + hl) * Kp;
        for (int k0 = 0; k0 < Kp; k0 += 32) {
            v16h av;
#pragma unroll
            for (int i = 0; i < 8; ++i) {
                av[i]     = (_Float16)(aBase[k0 + hi * 8 + i] * sc);
                av[8 + i] = (_Float16)(aBase[k0 + 16 + hi * 8 + i] * sc);
            }
#pragma unroll
            for (int j = 0; j < 4; ++j) {
                const _Float16* bB = (j == 0) ? bB0 : (j == 1) ? bB1 : (j == 2) ? bB2 : bB3;
                v8h b_lo = *(const v8h*)(bB + k0 + hi * 8);
                v8h b_hi = *(const v8h*)(bB + k0 + 16 + hi * 8);
                v16h bv;
#pragma unroll
                for (int i = 0; i < 8; ++i) { bv[i] = b_lo[i]; bv[8 + i] = b_hi[i]; }
                v8f& acc = (j == 0) ? acc0 : (j == 1) ? acc1 : (j == 2) ? acc2 : acc3;
                acc = __builtin_amdgcn_wmma_f32_16x16x32_f16(false, av, false, bv,
                                                             (short)0, acc, false, false);
            }
        }
    }

#pragma unroll
    for (int j = 0; j < 4; ++j) {
        v8f& acc = (j == 0) ? acc0 : (j == 1) ? acc1 : (j == 2) ? acc2 : acc3;
        int col = n0 + j * 16 + hl;
#pragma unroll
        for (int r = 0; r < 8; ++r) {
            int row = m0 + r + hi * 8;
            float v = acc[r];
            if (EPI == 0) {
                if (col < realN) {
                    v += bias[col];
                    v = (v >= 0.0f) ? v : 0.01f * v;
                    Cc[(size_t)row * Nout + col] = (_Float16)v;
                } else {
                    Cc[(size_t)row * Nout + col] = (_Float16)0.0f;
                }
            } else {
                v += bias[col];
                Cc[(size_t)row * Nout + col] = (_Float16)v;
                atomicAdd(&rownorm[row], v * v);
            }
        }
    }
}

// ---------------------------------------------------------------------------
// Graph-wise LN stats: one block per segment (segments are contiguous rows).
__global__ void k_ln_stats(const _Float16* __restrict__ x, const int* __restrict__ sep,
                           int stride, int realF, float* __restrict__ mean,
                           float* __restrict__ inv) {
    int s = blockIdx.x;
    int start = (s == 0) ? 0 : sep[s - 1];
    int end = sep[s];
    float s1 = 0.0f, s2 = 0.0f;
    for (int r = start; r < end; ++r)
        for (int c = threadIdx.x; c < realF; c += blockDim.x) {
            float v = (float)x[(size_t)r * stride + c];
            s1 += v; s2 += v * v;
        }
    __shared__ float sh1[256], sh2[256];
    sh1[threadIdx.x] = s1; sh2[threadIdx.x] = s2;
    __syncthreads();
    for (int off = 128; off > 0; off >>= 1) {
        if ((int)threadIdx.x < off) { sh1[threadIdx.x] += sh1[threadIdx.x + off];
                                      sh2[threadIdx.x] += sh2[threadIdx.x + off]; }
        __syncthreads();
    }
    if (threadIdx.x == 0) {
        float cnt = (float)((end - start) * realF);
        float m = sh1[0] / cnt;
        float var = sh2[0] / cnt - m * m;
        mean[s] = m;
        inv[s]  = rsqrtf(var + 1e-5f);
    }
}

__global__ void k_ln_apply(_Float16* __restrict__ x, const int* __restrict__ segid,
                           const float* __restrict__ mean, const float* __restrict__ inv,
                           const float* __restrict__ w, const float* __restrict__ b,
                           int stride, int realF) {
    int row = blockIdx.x;
    int c = blockIdx.y * blockDim.x + threadIdx.x;
    if (c >= realF) return;
    int s = segid[row];
    float v = (float)x[(size_t)row * stride + c];
    v = (v - mean[s]) * inv[s] * w[c] + b[c];
    x[(size_t)row * stride + c] = (_Float16)v;
}

// ---------------------------------------------------------------------------
__global__ void k_deg(const int* __restrict__ edges, int E, float* __restrict__ deg) {
    int e = blockIdx.x * blockDim.x + threadIdx.x;
    if (e >= E) return;
    atomicAdd(&deg[edges[E + e]], 1.0f);
}

__global__ void k_invdeg(const float* __restrict__ deg, float* __restrict__ invdeg, int N) {
    int i = blockIdx.x * blockDim.x + threadIdx.x;
    if (i >= N) return;
    invdeg[i] = 1.0f / fmaxf(deg[i], 1.0f);
}

// agg[dst] += x[src]  (f16 gather, f32 atomic accumulate), 4 cols per thread
__global__ void k_scatter(const _Float16* __restrict__ x, const int* __restrict__ edges,
                          int E, int dim, float* __restrict__ agg) {
    long long t = (long long)blockIdx.x * blockDim.x + threadIdx.x;
    int per = dim >> 2;
    if (t >= (long long)E * per) return;
    int e  = (int)(t / per);
    int c4 = (int)(t % per) * 4;
    int src = edges[e];
    int dst = edges[E + e];
#pragma unroll
    for (int j = 0; j < 4; ++j)
        atomicAdd(&agg[(size_t)dst * dim + c4 + j],
                  (float)x[(size_t)src * dim + c4 + j]);
}

// L2 row-normalize (+ optional ReLU), in place on f16 buffer
__global__ void k_l2norm(_Float16* __restrict__ x, const float* __restrict__ rownorm,
                         int dim, int relu) {
    int row = blockIdx.x;
    int c = blockIdx.y * blockDim.x + threadIdx.x;
    if (c >= dim) return;
    float sc = 1.0f / fmaxf(sqrtf(rownorm[row]), 1e-12f);
    float v = (float)x[(size_t)row * dim + c] * sc;
    if (relu) v = fmaxf(v, 0.0f);
    x[(size_t)row * dim + c] = (_Float16)v;
}

// segment sum over contiguous row ranges (no atomics)
__global__ void k_segsum(const _Float16* __restrict__ x, const int* __restrict__ sep,
                         float* __restrict__ segagg, int dim) {
    int s = blockIdx.x;
    int c = blockIdx.y * blockDim.x + threadIdx.x;
    if (c >= dim) return;
    int start = (s == 0) ? 0 : sep[s - 1];
    int end = sep[s];
    float acc = 0.0f;
    for (int i = start; i < end; ++i) acc += (float)x[(size_t)i * dim + c];
    segagg[(size_t)s * dim + c] = acc;
}

// out[s] = dot(segagg[s,:], Wf) + bf    (one wave32 per segment)
__global__ void k_final(const float* __restrict__ segagg, const float* __restrict__ Wf,
                        const float* __restrict__ bf, float* __restrict__ out, int dim) {
    int s = blockIdx.x;
    int lane = threadIdx.x;
    float acc = 0.0f;
    for (int c = lane; c < dim; c += 32) acc += segagg[(size_t)s * dim + c] * Wf[c];
    for (int off = 16; off > 0; off >>= 1) acc += __shfl_down(acc, off, 32);
    if (lane == 0) out[s] = acc + bf[0];
}

// ---------------------------------------------------------------------------
extern "C" void kernel_launch(void* const* d_in, const int* in_sizes, int n_in,
                              void* d_out, int out_size, void* d_ws, size_t ws_size,
                              hipStream_t stream) {
    const float* node_features = (const float*)d_in[0];
    const int*   node_cfg      = (const int*)d_in[1];
    const int*   sep           = (const int*)d_in[2];
    const int*   node_ops      = (const int*)d_in[3];
    const int*   edges         = (const int*)d_in[4];
    const float* opcode_emb    = (const float*)d_in[6];
    const float* cat_emb       = (const float*)d_in[7];
    const float* W1   = (const float*)d_in[8];
    const float* b1   = (const float*)d_in[9];
    const float* ln1w = (const float*)d_in[10];
    const float* ln1b = (const float*)d_in[11];
    const float* W2   = (const float*)d_in[12];
    const float* b2   = (const float*)d_in[13];
    const float* ln2w = (const float*)d_in[14];
    const float* ln2b = (const float*)d_in[15];
    const float* Wl0  = (const float*)d_in[16];
    const float* bl0  = (const float*)d_in[17];
    const float* Wr0  = (const float*)d_in[18];
    const float* Wl1  = (const float*)d_in[19];
    const float* bl1  = (const float*)d_in[20];
    const float* Wr1  = (const float*)d_in[21];
    const float* Wl2  = (const float*)d_in[22];
    const float* bl2  = (const float*)d_in[23];
    const float* Wr2  = (const float*)d_in[24];
    const float* Wf   = (const float*)d_in[25];
    const float* bfv  = (const float*)d_in[26];

    const int N = in_sizes[3];          // 65536
    const int S = in_sizes[2];          // 128
    const int E = in_sizes[4] / 2;      // 262144
    float* out = (float*)d_out;

    // workspace carve-up
    char* p = (char*)d_ws;
    auto alloc = [&](size_t bytes) { char* r = p; p += (bytes + 255) & ~(size_t)255; return r; };
    int*      segid   = (int*)alloc((size_t)N * 4);
    float*    deg     = (float*)alloc((size_t)N * 4);
    float*    invdeg  = (float*)alloc((size_t)N * 4);
    float*    rownorm = (float*)alloc((size_t)N * 4);
    float*    mean    = (float*)alloc((size_t)S * 4);
    float*    inv     = (float*)alloc((size_t)S * 4);
    float*    segagg  = (float*)alloc((size_t)S * 512 * 4);
    _Float16* W1T  = (_Float16*)alloc((size_t)384 * 192 * 2);
    _Float16* W2T  = (_Float16*)alloc((size_t)256 * 384 * 2);
    _Float16* Wl0T = (_Float16*)alloc((size_t)512 * 256 * 2);
    _Float16* Wr0T = (_Float16*)alloc((size_t)512 * 256 * 2);
    _Float16* Wl1T = (_Float16*)alloc((size_t)512 * 512 * 2);
    _Float16* Wr1T = (_Float16*)alloc((size_t)512 * 512 * 2);
    _Float16* Wl2T = (_Float16*)alloc((size_t)512 * 512 * 2);
    _Float16* Wr2T = (_Float16*)alloc((size_t)512 * 512 * 2);
    _Float16* bufA = (_Float16*)alloc((size_t)N * 512 * 2);
    _Float16* bufB = (_Float16*)alloc((size_t)N * 512 * 2);
    float*    agg  = (float*)alloc((size_t)N * 512 * 4);

    const int T = 256;
    auto cdiv = [](long long a, long long b) { return (int)((a + b - 1) / b); };

    hipMemsetAsync(deg, 0, (size_t)N * 4, stream);
    k_segid<<<cdiv(N, T), T, 0, stream>>>(sep, S, N, segid);

    // weight prep (transpose + pad + f16)
    k_convWT<<<cdiv(384 * 192, T), T, 0, stream>>>(W1,  W1T, 189, 378, 192, 384);
    k_convWT<<<cdiv(256 * 384, T), T, 0, stream>>>(W2,  W2T, 378, 256, 384, 256);
    k_convWT<<<cdiv(512 * 256, T), T, 0, stream>>>(Wl0, Wl0T, 256, 512, 256, 512);
    k_convWT<<<cdiv(512 * 256, T), T, 0, stream>>>(Wr0, Wr0T, 256, 512, 256, 512);
    k_convWT<<<cdiv(512 * 512, T), T, 0, stream>>>(Wl1, Wl1T, 512, 512, 512, 512);
    k_convWT<<<cdiv(512 * 512, T), T, 0, stream>>>(Wr1, Wr1T, 512, 512, 512, 512);
    k_convWT<<<cdiv(512 * 512, T), T, 0, stream>>>(Wl2, Wl2T, 512, 512, 512, 512);
    k_convWT<<<cdiv(512 * 512, T), T, 0, stream>>>(Wr2, Wr2T, 512, 512, 512, 512);

    // feature assembly
    k_build_x0<<<cdiv((long long)N * 192, T), T, 0, stream>>>(
        node_features, node_cfg, node_ops, opcode_emb, cat_emb, bufA, N);

    // MLP stage 1: [N,192] @ [192,384] -> leaky -> graphwise LN
    gemm_wmma<0><<<dim3(N / 128, 384 / 64), T, 0, stream>>>(
        bufA, W1T, nullptr, nullptr, nullptr, b1, bufB, nullptr, N, 192, 384, 378);
    k_ln_stats<<<S, T, 0, stream>>>(bufB, sep, 384, 378, mean, inv);
    k_ln_apply<<<dim3(N, cdiv(378, T)), T, 0, stream>>>(bufB, segid, mean, inv, ln1w, ln1b, 384, 378);

    // MLP stage 2: [N,384] @ [384,256] -> leaky -> graphwise LN
    gemm_wmma<0><<<dim3(N / 128, 256 / 64), T, 0, stream>>>(
        bufB, W2T, nullptr, nullptr, nullptr, b2, bufA, nullptr, N, 384, 256, 256);
    k_ln_stats<<<S, T, 0, stream>>>(bufA, sep, 256, 256, mean, inv);
    k_ln_apply<<<dim3(N, cdiv(256, T)), T, 0, stream>>>(bufA, segid, mean, inv, ln2w, ln2b, 256, 256);

    // degree (constant across layers)
    k_deg<<<cdiv(E, T), T, 0, stream>>>(edges, E, deg);
    k_invdeg<<<cdiv(N, T), T, 0, stream>>>(deg, invdeg, N);

    // SAGE layer 0: 256 -> 512, ReLU
    hipMemsetAsync(agg, 0, (size_t)N * 256 * 4, stream);
    k_scatter<<<cdiv((long long)E * 64, T), T, 0, stream>>>(bufA, edges, E, 256, agg);
    hipMemsetAsync(rownorm, 0, (size_t)N * 4, stream);
    gemm_wmma<1><<<dim3(N / 128, 512 / 64), T, 0, stream>>>(
        bufA, Wr0T, agg, Wl0T, invdeg, bl0, bufB, rownorm, N, 256, 512, 512);
    k_l2norm<<<dim3(N, 2), T, 0, stream>>>(bufB, rownorm, 512, 1);

    // SAGE layer 1: 512 -> 512, ReLU
    hipMemsetAsync(agg, 0, (size_t)N * 512 * 4, stream);
    k_scatter<<<cdiv((long long)E * 128, T), T, 0, stream>>>(bufB, edges, E, 512, agg);
    hipMemsetAsync(rownorm, 0, (size_t)N * 4, stream);
    gemm_wmma<1><<<dim3(N / 128, 512 / 64), T, 0, stream>>>(
        bufB, Wr1T, agg, Wl1T, invdeg, bl1, bufA, rownorm, N, 512, 512, 512);
    k_l2norm<<<dim3(N, 2), T, 0, stream>>>(bufA, rownorm, 512, 1);

    // SAGE layer 2: 512 -> 512, no ReLU
    hipMemsetAsync(agg, 0, (size_t)N * 512 * 4, stream);
    k_scatter<<<cdiv((long long)E * 128, T), T, 0, stream>>>(bufA, edges, E, 512, agg);
    hipMemsetAsync(rownorm, 0, (size_t)N * 4, stream);
    gemm_wmma<1><<<dim3(N / 128, 512 / 64), T, 0, stream>>>(
        bufA, Wr2T, agg, Wl2T, invdeg, bl2, bufB, rownorm, N, 512, 512, 512);
    k_l2norm<<<dim3(N, 2), T, 0, stream>>>(bufB, rownorm, 512, 0);

    // readout: segment sum + linear head -> 128 outputs
    k_segsum<<<dim3(S, 2), T, 0, stream>>>(bufB, sep, segagg, 512);
    k_final<<<S, 32, 0, stream>>>(segagg, Wf, bfv, out, 512);
    (void)n_in; (void)out_size; (void)ws_size;
}
